// HistogramLoss_2645699854669
// MI455X (gfx1250) — matrix operand order repaired and checked by reading, not compiled, not run on hardware
//
#include <hip/hip_runtime.h>
#include <hip/hip_bf16.h>

typedef __attribute__((ext_vector_type(16))) _Float16 v16h;
typedef __attribute__((ext_vector_type(8)))  float    v8f;
typedef __attribute__((ext_vector_type(4)))  float    v4f;
typedef __attribute__((ext_vector_type(4)))  int      v4i;

#define MAXB 256          // max bins supported by LDS staging
#define MAXE (MAXB + 1)

// ---------------------------------------------------------------------------
// Zero the workspace counters (harness poisons ws with 0xAA; we must clear
// every launch since the graph replays this exact sequence).
// ---------------------------------------------------------------------------
__global__ void HL_zero_kernel(unsigned int* __restrict__ cnt, int n) {
    int i = blockIdx.x * blockDim.x + threadIdx.x;
    if (i < n) cnt[i] = 0u;
}

// ---------------------------------------------------------------------------
// Dual masked histogram: one pass streams obs+pred+mask (mask read once,
// feeds both histograms). Per-block LDS privatization (ds_add), single
// global integer-atomic flush per block -> deterministic counts.
// searchsorted(edges, x, side='right')-1 semantics via branchless binary
// search over edges staged in LDS; x == edges[nb] goes to the last bin.
// ---------------------------------------------------------------------------
__global__ __launch_bounds__(256) void HL_hist_kernel(
    const float* __restrict__ obs, const float* __restrict__ pred,
    const int* __restrict__ mask, const float* __restrict__ edges,
    unsigned int* __restrict__ cnt_obs, unsigned int* __restrict__ cnt_pred,
    int HW, int nb) {
    __shared__ float se[MAXE];
    __shared__ unsigned int h[2 * MAXB];

    const int tid = threadIdx.x;
    const int ne  = nb + 1;
    for (int i = tid; i < ne; i += blockDim.x) se[i] = edges[i];
    for (int i = tid; i < 2 * nb; i += blockDim.x) h[i] = 0u;
    __syncthreads();

    const int b = blockIdx.y;
    const long long base = (long long)b * HW;
    const float* ob = obs  + base;
    const float* pr = pred + base;
    const int*   mk = mask + base;
    const float  etop = se[nb];
    const int nquads = (HW + 3) >> 2;

    for (int q = blockIdx.x * blockDim.x + tid; q < nquads;
         q += gridDim.x * blockDim.x) {
        const int i0 = q << 2;
        v4f xo, xp; v4i mm;
        if (i0 + 3 < HW) {
            // 16B non-temporal streaming loads: data is touched exactly once
            // and the working set (~400MB) exceeds the 192MB L2.
            xo = __builtin_nontemporal_load((const v4f*)(ob + i0));
            xp = __builtin_nontemporal_load((const v4f*)(pr + i0));
            mm = __builtin_nontemporal_load((const v4i*)(mk + i0));
        } else {
            xo = (v4f)0.0f; xp = (v4f)0.0f; mm = (v4i)0;
            for (int e = 0; e < 4; ++e)
                if (i0 + e < HW) { xo[e] = ob[i0+e]; xp[e] = pr[i0+e]; mm[e] = mk[i0+e]; }
        }
        #pragma unroll
        for (int e = 0; e < 4; ++e) {
            if (i0 + e < HW && mm[e]) {
                {   // obs
                    const float x = xo[e];
                    int lo = 0, hi = ne;          // count of edges <= x
                    while (lo < hi) { int mid = (lo + hi) >> 1;
                                      if (se[mid] <= x) lo = mid + 1; else hi = mid; }
                    int idx = lo - 1;
                    if (x == etop) idx = nb - 1;
                    if (idx >= 0 && idx < nb) atomicAdd(&h[idx], 1u);
                }
                {   // pred
                    const float x = xp[e];
                    int lo = 0, hi = ne;
                    while (lo < hi) { int mid = (lo + hi) >> 1;
                                      if (se[mid] <= x) lo = mid + 1; else hi = mid; }
                    int idx = lo - 1;
                    if (x == etop) idx = nb - 1;
                    if (idx >= 0 && idx < nb) atomicAdd(&h[nb + idx], 1u);
                }
            }
        }
    }
    __syncthreads();
    for (int i = tid; i < nb; i += blockDim.x) {
        unsigned int vo = h[i], vp = h[nb + i];
        if (vo) atomicAdd(&cnt_obs [(size_t)b * nb + i], vo);
        if (vp) atomicAdd(&cnt_pred[(size_t)b * nb + i], vp);
    }
}

// ---------------------------------------------------------------------------
// Finalize (single wave32, EXEC all-ones as WMMA requires):
//  1) proportions + empty-sample fallback, write p_obs / p_pred to d_out
//  2) cdf_obs - cdf_pred = cumsum(p_obs - p_pred) computed as the matmul
//     D(Bx64,f16 diffs) x U(64x64 upper-triangular ones) with
//     v_wmma_f32_16x16x32_f16 (f32 accumulate), U generated in registers.
//  3) rarity-weighted W2 reduce via shfl_xor, lane0 writes the scalar loss.
// ---------------------------------------------------------------------------
__global__ __launch_bounds__(32) void HL_finalize_kernel(
    const unsigned int* __restrict__ cnt_obs,
    const unsigned int* __restrict__ cnt_pred,
    const float* __restrict__ edges, const float* __restrict__ weights,
    float* __restrict__ out, int B, int nb) {
    __shared__ _Float16 diff[64 * 64];   // row-major [b][bin], valid for nb<=64
    __shared__ float se[MAXE];

    const int lane = threadIdx.x;
    for (int i = lane; i < nb + 1; i += 32) se[i] = edges[i];

    // --- proportions + f16 diff staging (one row per lane-iteration) -------
    for (int b = lane; b < B; b += 32) {
        unsigned int to = 0, tp = 0;
        for (int i = 0; i < nb; ++i) { to += cnt_obs[b*nb + i]; tp += cnt_pred[b*nb + i]; }
        const float fallback = 1.0f / (float)nb;     // counts==all-ones path
        const float inv_o = (to != 0u) ? 1.0f / (float)to : 0.0f;
        const float inv_p = (tp != 0u) ? 1.0f / (float)tp : 0.0f;
        for (int i = 0; i < nb; ++i) {
            float po = (to != 0u) ? (float)cnt_obs [b*nb + i] * inv_o : fallback;
            float pp = (tp != 0u) ? (float)cnt_pred[b*nb + i] * inv_p : fallback;
            out[1 + b*nb + i]             = po;
            out[1 + B*nb + b*nb + i]      = pp;
            if (nb <= 64) diff[b*64 + i]  = (_Float16)(po - pp);
        }
    }
    __syncthreads();

    float local = 0.0f;
    const int mloc = lane & 15;        // N (or M) within a 16-wide tile
    const int half = lane >> 4;        // half-wave selector per ISA layouts

    if (nb == 64 && (B & 15) == 0 && B <= 64) {
        // --- WMMA path: cumsum as triangular matmul ------------------------
        const int mtiles = B >> 4;
        for (int mt = 0; mt < mtiles; ++mt) {
            for (int ct = 0; ct < 4; ++ct) {
                v8f acc = {};
                #pragma unroll
                for (int kc = 0; kc < 2; ++kc) {
                    v16h a, bm;
                    #pragma unroll
                    for (int v = 0; v < 8; ++v) {
                        // A 16x32 f16 layout (ISA 7.12.2): lane M=mloc,
                        // VGPR v packs K = 2v (+8 if v>=4) + 8*half, K+1
                        const int kb  = 2*v + ((v >= 4) ? 8 : 0) + 8*half + 32*kc;
                        const int row = 16*mt + mloc;
                        a[2*v]     = diff[row*64 + kb];
                        a[2*v + 1] = diff[row*64 + kb + 1];
                        // B 32x16 f16 layout: lane N=mloc, VGPR v packs
                        // K = 2v (+8 if v>=4) + 8*half, K+1 ; U[k][n] = (k <= n)
                        const int kl = 2*v + ((v >= 4) ? 8 : 0) + 8*half + 32*kc;
                        const int gn = 16*ct + mloc;
                        bm[2*v]     = (kl     <= gn) ? (_Float16)1.0f : (_Float16)0.0f;
                        bm[2*v + 1] = (kl + 1 <= gn) ? (_Float16)1.0f : (_Float16)0.0f;
                    }
                    acc = __builtin_amdgcn_wmma_f32_16x16x32_f16(
                        false, a, false, bm, (short)0, acc, false, false);
                }
                // D layout: VGPR r -> M = r + 8*half; N = mloc
                const int gj = 16*ct + mloc;
                int jj = (gj < nb - 1) ? gj : (nb - 2);
                if (jj < 0) jj = 0;
                const float wj = 0.5f * (se[jj + 2] - se[jj]) * weights[gj];
                #pragma unroll
                for (int r = 0; r < 8; ++r) { float c = acc[r]; local += c * c * wj; }
            }
        }
    } else {
        // --- scalar fallback for unusual shapes ----------------------------
        for (int b = lane; b < B; b += 32) {
            float cdf = 0.0f;
            for (int i = 0; i < nb; ++i) {
                float d = out[1 + b*nb + i] - out[1 + B*nb + b*nb + i];
                cdf += d;
                int jj = (i < nb - 1) ? i : (nb - 2);
                if (jj < 0) jj = 0;
                float w = 0.5f * (se[jj + 2] - se[jj]) * weights[i];
                local += cdf * cdf * w;
            }
        }
    }

    // wave32 reduction
    #pragma unroll
    for (int off = 16; off > 0; off >>= 1)
        local += __shfl_xor(local, off, 32);
    if (lane == 0) out[0] = local / (float)B;
}

// ---------------------------------------------------------------------------
extern "C" void kernel_launch(void* const* d_in, const int* in_sizes, int n_in,
                              void* d_out, int out_size, void* d_ws, size_t ws_size,
                              hipStream_t stream) {
    const float* obs     = (const float*)d_in[0];
    const float* pred    = (const float*)d_in[1];
    const int*   mask    = (const int*)d_in[2];
    const float* edges   = (const float*)d_in[3];
    const float* weights = (const float*)d_in[4];
    float* out = (float*)d_out;

    const int N  = in_sizes[0];
    const int nb = in_sizes[4];                 // bin_weights length
    const int B  = (out_size - 1) / (2 * nb);   // out = 1 + 2*B*nb
    const int HW = N / B;

    unsigned int* cnt_obs  = (unsigned int*)d_ws;
    unsigned int* cnt_pred = cnt_obs + (size_t)B * nb;

    const int ncnt = 2 * B * nb;
    HL_zero_kernel<<<(ncnt + 255) / 256, 256, 0, stream>>>(cnt_obs, ncnt);

    // ~128 blocks per batch: plenty of waves for occupancy while keeping the
    // global atomic flush tiny (4096 blocks * 128 atomics ~= 0.5M).
    const int nquads = (HW + 3) / 4;
    int bpb = (nquads + 255) / 256;
    if (bpb > 128) bpb = 128;
    dim3 grid(bpb, B);
    HL_hist_kernel<<<grid, 256, 0, stream>>>(obs, pred, mask, edges,
                                             cnt_obs, cnt_pred, HW, nb);

    HL_finalize_kernel<<<1, 32, 0, stream>>>(cnt_obs, cnt_pred, edges, weights,
                                             out, B, nb);
}
